// xLSTMBlock_10531259810152
// MI455X (gfx1250) — matrix operand order repaired and checked by reading, not compiled, not run on hardware
//
#include <hip/hip_runtime.h>
#include <hip/hip_bf16.h>
#include <math.h>

#define BSZ  8
#define SEQ  2048
#define HID  1024
#define MTOT (BSZ * SEQ)        // 16384
#define NBLK_SCAN 32u

typedef __bf16 bf16;
typedef bf16  v16bf __attribute__((ext_vector_type(16)));
typedef bf16  v8bf  __attribute__((ext_vector_type(8)));
typedef float v8f   __attribute__((ext_vector_type(8)));

// gfx1250 async global->LDS path (ASYNCcnt-tracked), guarded so the file
// compiles on toolchains without the builtins.
#if defined(__has_builtin)
#if __has_builtin(__builtin_amdgcn_global_load_async_to_lds_b128) && \
    __has_builtin(__builtin_amdgcn_s_wait_asynccnt)
#define HAVE_ASYNC_LDS 1
#endif
#endif

#if defined(HAVE_ASYNC_LDS)
// Builtin signature (from hipcc diagnostic): param0 = int4-vector pointer in
// the global (__device__/AS1) address space, param1 = LDS (AS3) pointer,
// then imm offset and imm cpol.
typedef int v4i_gcc __attribute__((vector_size(16)));
typedef __attribute__((address_space(1))) v4i_gcc* as1_v4i_ptr;
typedef __attribute__((address_space(3))) v4i_gcc* as3_v4i_ptr;
#endif

// ---------------- WMMA helpers (contiguous k order: k = 16*(lane/16) + s) ----------------

__device__ __forceinline__ v8f wmma_bf16(v16bf a, v16bf b, v8f c) {
  return __builtin_amdgcn_wmma_f32_16x16x32_bf16(false, a, false, b,
                                                 (short)0, c, false, false);
}

// A fragment: 16x32 bf16, row-major source. lane<16: M=lane, K=0..15;
// lane>=16: M=lane-16, K=16..31. base points at (row0, k0).
__device__ __forceinline__ v16bf load_a16(const bf16* base, int ld, int lane) {
  int m  = lane & 15;
  int ko = (lane >> 4) << 4;
  const bf16* p = base + m * ld + ko;
  v8bf lo = *(const v8bf*)(p);
  v8bf hi = *(const v8bf*)(p + 8);
  v16bf a;
#pragma unroll
  for (int e = 0; e < 8; ++e) { a[e] = lo[e]; a[e + 8] = hi[e]; }
  return a;
}

// B fragment: 32x16 bf16 from W^T stored [n][k] (k contiguous, ld=1024).
// lane<16: N=lane, K=0..15 ; lane>=16: N=lane-16, K=16..31.
__device__ __forceinline__ v16bf load_b16_global(const bf16* wt_col0, int k0, int lane) {
  int n  = lane & 15;
  int kb = k0 + ((lane >> 4) << 4);
  const bf16* p = wt_col0 + n * HID + kb;
  v8bf lo = *(const v8bf*)(p);
  v8bf hi = *(const v8bf*)(p + 8);
  v16bf b;
#pragma unroll
  for (int e = 0; e < 8; ++e) { b[e] = lo[e]; b[e + 8] = hi[e]; }
  return b;
}

// B fragment from LDS. Column cL stored as 128 16B units; unit u lives at
// physical unit (u ^ ((cL&7)<<1)) -> 16 lanes hit 8 distinct bank groups.
__device__ __forceinline__ v16bf load_b16_lds(const bf16* sm, int cbase, int k0, int lane) {
  int cL = cbase + (lane & 15);
  int u0 = (k0 >> 3) + ((lane >> 4) << 1);       // even
  int sw = (cL & 7) << 1;                        // even
  const bf16* p = sm + cL * HID + ((u0 ^ sw) << 3);
  v8bf lo = *(const v8bf*)(p);
  v8bf hi = *(const v8bf*)(p + 8);               // (u0+1)^sw == (u0^sw)+1
  v16bf b;
#pragma unroll
  for (int e = 0; e < 8; ++e) { b[e] = lo[e]; b[e + 8] = hi[e]; }
  return b;
}

// ---------------- generation grid barrier ----------------

__device__ __forceinline__ void grid_sync(unsigned* sync, unsigned nblk) {
  __syncthreads();
  if (threadIdx.x == 0) {
    __threadfence();
    unsigned gen = __atomic_load_n(&sync[1], __ATOMIC_ACQUIRE);
    unsigned arrived = __atomic_add_fetch(&sync[0], 1u, __ATOMIC_ACQ_REL);
    if (arrived == nblk) {
      __atomic_store_n(&sync[0], 0u, __ATOMIC_RELAXED);
      __atomic_add_fetch(&sync[1], 1u, __ATOMIC_ACQ_REL);
    } else {
      while (__atomic_load_n(&sync[1], __ATOMIC_ACQUIRE) == gen) {
        __builtin_amdgcn_s_sleep(2);
      }
    }
    __threadfence();
  }
  __syncthreads();
}

// ---------------- prep kernels ----------------

__global__ void k_cvt(const float* __restrict__ src, bf16* __restrict__ dst, int n) {
  int i = blockIdx.x * blockDim.x + threadIdx.x;
  int s = gridDim.x * blockDim.x;
  for (; i < n; i += s) dst[i] = (bf16)src[i];
}

// dst[h*1024+k] = src[k*1024+h]  (1024x1024 transpose + fp32->bf16)
__global__ void k_tcvt(const float* __restrict__ src, bf16* __restrict__ dst) {
  int i = blockIdx.x * blockDim.x + threadIdx.x;
  int s = gridDim.x * blockDim.x;
  for (; i < HID * HID; i += s) {
    int h = i >> 10, k = i & 1023;
    dst[i] = (bf16)src[k * HID + h];
  }
}

__global__ void k_init(unsigned* hbf, unsigned* gbf, unsigned* cbuf,
                       unsigned* nbuf, unsigned* sync) {
  int i = blockIdx.x * 256 + threadIdx.x;
  int s = gridDim.x * 256;
  for (int j = i; j < 8192; j += s) { hbf[j] = 0u; gbf[j] = 0u; cbuf[j] = 0u; nbuf[j] = 0u; }
  if (blockIdx.x == 0 && threadIdx.x < 64) sync[threadIdx.x] = 0u;
}

// ---------------- pre-projection GEMM:  pre[g][m][h] = xb @ Wgx^T + bias ----------------
// C tile 128x128 per block; 8 waves (2M x 4N); wave tile 64x32 = 4x2 WMMA tiles.

__global__ __launch_bounds__(256)
void k_gemm_pre(const bf16* __restrict__ xb, const bf16* __restrict__ wxt,
                const float* __restrict__ bi, const float* __restrict__ bfv,
                const float* __restrict__ bg, const float* __restrict__ bo,
                bf16* __restrict__ pre) {
  const int lane = threadIdx.x & 31, wave = threadIdx.x >> 5;
  const int bm = blockIdx.x & 127, bn = blockIdx.x >> 7;   // 128 x 32 blocks
  const int wm = wave & 1, wn = wave >> 1;
  const int row0 = bm * 128 + wm * 64;
  const int col0 = bn * 128 + wn * 32;                     // global col in [0,4096)
  const int gate = col0 >> 10;
  const int h0   = col0 & 1023;
  const bf16* wt = wxt + (size_t)gate * HID * HID;
  const float* bias = (gate == 0) ? bi : (gate == 1) ? bfv : (gate == 2) ? bg : bo;

  v8f acc[4][2];
#pragma unroll
  for (int a = 0; a < 4; ++a)
#pragma unroll
    for (int b = 0; b < 2; ++b)
      acc[a][b] = (v8f){0.f,0.f,0.f,0.f,0.f,0.f,0.f,0.f};

  for (int k0 = 0; k0 < HID; k0 += 32) {
    v16bf afr[4];
#pragma unroll
    for (int mi = 0; mi < 4; ++mi)
      afr[mi] = load_a16(xb + (size_t)(row0 + mi * 16) * HID + k0, HID, lane);
    v16bf bfr[2];
#pragma unroll
    for (int ni = 0; ni < 2; ++ni)
      bfr[ni] = load_b16_global(wt + (size_t)(h0 + ni * 16) * HID, k0, lane);
#pragma unroll
    for (int mi = 0; mi < 4; ++mi)
#pragma unroll
      for (int ni = 0; ni < 2; ++ni)
        acc[mi][ni] = wmma_bf16(afr[mi], bfr[ni], acc[mi][ni]);
  }

  const int mofs = (lane >> 4) << 3;
  const int nl   = lane & 15;
  bf16* preg = pre + (size_t)gate * MTOT * HID;
#pragma unroll
  for (int mi = 0; mi < 4; ++mi)
#pragma unroll
    for (int ni = 0; ni < 2; ++ni) {
      int h = h0 + ni * 16 + nl;
      float bv = bias[h];
#pragma unroll
      for (int r = 0; r < 8; ++r) {
        int m = row0 + mi * 16 + mofs + r;
        preg[(size_t)m * HID + h] = (bf16)(acc[mi][ni][r] + bv);
      }
    }
}

// ---------------- persistent recurrent scan ----------------
// 32 blocks x 256 threads (8 waves). LDS (320KB): 128 gate cols + 32 Weg cols,
// each 1024 bf16 (2KB), swizzled. Phase A: h@[Wih|Wfh|Wgh|Woh] + activations.
// Phase B: g@Weg, state update, write h_t.

__global__ __launch_bounds__(256)
void k_scan(const bf16* __restrict__ wht, const bf16* __restrict__ wegt,
            const bf16* __restrict__ pre, float* __restrict__ gact,
            bf16* __restrict__ gbf, bf16* __restrict__ hbf,
            float* __restrict__ cbuf, float* __restrict__ nbuf,
            const float* __restrict__ beg, float* __restrict__ dout,
            unsigned* __restrict__ sync) {
  extern __shared__ bf16 sm[];            // 160 * 1024 bf16 = 320KB
  const int tid = threadIdx.x, lane = tid & 31, wv = tid >> 5;
  const int blk = blockIdx.x;

  // Stage this WGP's weight columns into LDS (once). With the gfx1250 async
  // engine this is ASYNCcnt-tracked global_load_async_to_lds_b128; otherwise
  // a plain global->VGPR->ds_store copy.
  for (int u = tid; u < 160 * 128; u += 256) {
    int cL = u >> 7, un = u & 127;
    const bf16* src;
    if (cL < 128) {
      int w    = cL >> 4;
      int ct   = blk * 8 + w;             // 0..255 gate column tiles
      int gate = ct >> 6;
      int hg   = ((ct & 63) << 4) + (cL & 15);
      src = wht + (size_t)gate * HID * HID + (size_t)hg * HID;
    } else {
      int j   = cL - 128;
      int ct2 = blk * 2 + (j >> 4);       // 0..63 Weg column tiles
      int hg  = (ct2 << 4) + (j & 15);
      src = wegt + (size_t)hg * HID;
    }
    int sw = (cL & 7) << 1;
    bf16* dst = sm + cL * HID + ((un ^ sw) << 3);
#if defined(HAVE_ASYNC_LDS)
    __builtin_amdgcn_global_load_async_to_lds_b128(
        (as1_v4i_ptr)(src + un * 8), (as3_v4i_ptr)dst, 0, 0);
#else
    uint4 d = *(const uint4*)(src + un * 8);
    *(uint4*)dst = d;
#endif
  }
#if defined(HAVE_ASYNC_LDS)
  __builtin_amdgcn_s_wait_asynccnt(0);
#endif
  __syncthreads();

  const int ct   = blk * 8 + wv;          // this wave's gate tile
  const int gate = ct >> 6;
  const int hb   = (ct & 63) << 4;
  const int cb   = wv << 4;               // LDS col base (phase A)
  const int mofs = (lane >> 4) << 3;
  const int nl   = lane & 15;
  const bf16* preg = pre + (size_t)gate * MTOT * HID;

  for (int t = 0; t < SEQ; ++t) {
    // Pull this step's gate preactivation line toward the WGP (global_prefetch_b8).
    __builtin_prefetch((const void*)(preg + (size_t)((lane >> 4) * SEQ + t) * HID + hb + nl), 0, 1);

    // ---- phase A: preact = h @ Wh, activations ----
    v8f acc = (v8f){0.f,0.f,0.f,0.f,0.f,0.f,0.f,0.f};
    for (int k0 = 0; k0 < HID; k0 += 32) {
      v16bf a = load_a16(hbf + k0, HID, lane);
      v16bf b = load_b16_lds(sm, cb, k0, lane);
      acc = wmma_bf16(a, b, acc);
    }
#pragma unroll
    for (int r = 0; r < 8; ++r) {
      int m = mofs + r;                    // batch 0..15 (8..15 are padding)
      if (m < BSZ) {
        int n = hb + nl;
        float v = acc[r] + (float)preg[(size_t)(m * SEQ + t) * HID + n];
        float act = (gate == 2) ? tanhf(v) : 1.f / (1.f + __expf(-v));
        gact[gate * 8192 + m * HID + n] = act;
        if (gate == 2) gbf[m * HID + n] = (bf16)act;
      }
    }
    grid_sync(sync, NBLK_SCAN);

    // ---- phase B: eg = g @ Weg, state update ----
    if (wv < 2) {
      int ct2 = blk * 2 + wv;
      int h2  = ct2 << 4;
      v8f eg = (v8f){0.f,0.f,0.f,0.f,0.f,0.f,0.f,0.f};
      for (int k0 = 0; k0 < HID; k0 += 32) {
        v16bf a = load_a16(gbf + k0, HID, lane);
        v16bf b = load_b16_lds(sm, 128 + (wv << 4), k0, lane);
        eg = wmma_bf16(a, b, eg);
      }
#pragma unroll
      for (int r = 0; r < 8; ++r) {
        int m = mofs + r;
        if (m < BSZ) {
          int n  = h2 + nl;
          float i_ = gact[0 * 8192 + m * HID + n];
          float f_ = gact[1 * 8192 + m * HID + n];
          float g_ = gact[2 * 8192 + m * HID + n];
          float o_ = gact[3 * 8192 + m * HID + n];
          float cn = f_ * cbuf[m * HID + n] + i_ * g_;
          float nn = f_ * nbuf[m * HID + n] + i_ * __expf(eg[r] + beg[n]);
          float hn = o_ * (cn / nn);
          cbuf[m * HID + n] = cn;
          nbuf[m * HID + n] = nn;
          hbf[m * HID + n]  = (bf16)hn;
          dout[(size_t)(m * SEQ + t) * HID + n] = hn;
          if (t == SEQ - 1) {
            size_t tail = (size_t)MTOT * HID;
            dout[tail +          m * HID + n] = hn;
            dout[tail +  8192 +  m * HID + n] = cn;
            dout[tail + 16384 +  m * HID + n] = nn;
          }
        }
      }
    }
    grid_sync(sync, NBLK_SCAN);
  }
}

// ---------------- fused residual + LayerNorm (in place on d_out) ----------------

__global__ __launch_bounds__(256)
void k_ln(const float* __restrict__ x, const float* __restrict__ gamma,
          const float* __restrict__ beta, float* __restrict__ out) {
  __shared__ float rs[256], rs2[256];
  size_t row = blockIdx.x;
  const float* xr = x + row * HID;
  float* orow = out + row * HID;
  float v[4], s = 0.f, s2 = 0.f;
#pragma unroll
  for (int j = 0; j < 4; ++j) {
    int c = threadIdx.x + j * 256;
    v[j] = orow[c] + xr[c];
    s += v[j]; s2 += v[j] * v[j];
  }
  rs[threadIdx.x] = s; rs2[threadIdx.x] = s2;
  __syncthreads();
  for (int off = 128; off > 0; off >>= 1) {
    if (threadIdx.x < off) {
      rs[threadIdx.x]  += rs[threadIdx.x + off];
      rs2[threadIdx.x] += rs2[threadIdx.x + off];
    }
    __syncthreads();
  }
  float mu  = rs[0] * (1.f / HID);
  float var = rs2[0] * (1.f / HID) - mu * mu;
  float inv = rsqrtf(var + 1e-5f);
#pragma unroll
  for (int j = 0; j < 4; ++j) {
    int c = threadIdx.x + j * 256;
    orow[c] = (v[j] - mu) * inv * gamma[c] + beta[c];
  }
}

// ---------------- host launch ----------------

extern "C" void kernel_launch(void* const* d_in, const int* in_sizes, int n_in,
                              void* d_out, int out_size, void* d_ws, size_t ws_size,
                              hipStream_t stream) {
  (void)in_sizes; (void)n_in; (void)out_size; (void)ws_size;
  const float* x    = (const float*)d_in[0];
  const float* Wi   = (const float*)d_in[1];
  const float* bi   = (const float*)d_in[2];
  const float* Wf   = (const float*)d_in[3];
  const float* bfv  = (const float*)d_in[4];
  const float* Wg   = (const float*)d_in[5];
  const float* bg   = (const float*)d_in[6];
  const float* Wo   = (const float*)d_in[7];
  const float* bo   = (const float*)d_in[8];
  const float* Weg  = (const float*)d_in[9];
  const float* beg  = (const float*)d_in[10];
  const float* gamma= (const float*)d_in[11];
  const float* beta = (const float*)d_in[12];
  float* out = (float*)d_out;

  char* ws = (char*)d_ws;
  size_t o = 0;
  auto alloc = [&](size_t bytes) -> char* {
    char* p = ws + o; o += (bytes + 255) & ~(size_t)255; return p;
  };
  bf16*  xb   = (bf16*) alloc((size_t)MTOT * HID * 2);          // x in bf16
  bf16*  wxt  = (bf16*) alloc((size_t)4 * HID * HID * 2);       // W{ifgo}x^T
  bf16*  wht  = (bf16*) alloc((size_t)4 * HID * HID * 2);       // W{ifgo}h^T
  bf16*  wegt = (bf16*) alloc((size_t)HID * HID * 2);           // Weg^T
  bf16*  pre  = (bf16*) alloc((size_t)4 * MTOT * HID * 2);      // gate preacts
  float* gact = (float*)alloc((size_t)4 * BSZ * HID * 4);       // activated gates
  bf16*  gbf  = (bf16*) alloc((size_t)16 * HID * 2);            // g (padded to 16 rows)
  bf16*  hbf  = (bf16*) alloc((size_t)16 * HID * 2);            // h (padded to 16 rows)
  float* cbuf = (float*)alloc((size_t)BSZ * HID * 4);
  float* nbuf = (float*)alloc((size_t)BSZ * HID * 4);
  unsigned* sync = (unsigned*)alloc(256);

  (void)hipFuncSetAttribute((const void*)k_scan,
                            hipFuncAttributeMaxDynamicSharedMemorySize, 327680);

  k_cvt<<<4096, 256, 0, stream>>>(x, xb, MTOT * HID);
  k_tcvt<<<1024, 256, 0, stream>>>(Wi,               wxt + 0 * 1048576);
  k_tcvt<<<1024, 256, 0, stream>>>(Wf,               wxt + 1 * 1048576);
  k_tcvt<<<1024, 256, 0, stream>>>(Wg,               wxt + 2 * 1048576);
  k_tcvt<<<1024, 256, 0, stream>>>(Wo,               wxt + 3 * 1048576);
  k_tcvt<<<1024, 256, 0, stream>>>(Wi + 1048576,     wht + 0 * 1048576);
  k_tcvt<<<1024, 256, 0, stream>>>(Wf + 1048576,     wht + 1 * 1048576);
  k_tcvt<<<1024, 256, 0, stream>>>(Wg + 1048576,     wht + 2 * 1048576);
  k_tcvt<<<1024, 256, 0, stream>>>(Wo + 1048576,     wht + 3 * 1048576);
  k_tcvt<<<1024, 256, 0, stream>>>(Weg,              wegt);
  k_init<<<64, 256, 0, stream>>>((unsigned*)hbf, (unsigned*)gbf,
                                 (unsigned*)cbuf, (unsigned*)nbuf, sync);
  k_gemm_pre<<<4096, 256, 0, stream>>>(xb, wxt, bi, bfv, bg, bo, pre);
  k_scan<<<NBLK_SCAN, 256, 327680, stream>>>(wht, wegt, pre, gact, gbf, hbf,
                                             cbuf, nbuf, beg, out, sync);
  k_ln<<<MTOT, 256, 0, stream>>>(x, gamma, beta, out);
}
